// Attention_77498389889672
// MI455X (gfx1250) — matrix operand order repaired and checked by reading, not compiled, not run on hardware
//
#include <hip/hip_runtime.h>

// MI455X (gfx1250) attention: bf16 WMMA GEMMs + flash attention.
// GEMM A-tiles are staged via the Tensor Data Mover (tensor_load_to_lds,
// TENSORcnt double-buffered); all matmuls lower to v_wmma_f32_16x16x32_bf16.

typedef __attribute__((ext_vector_type(16))) __bf16 v16bf;
typedef __attribute__((ext_vector_type(8)))  __bf16 v8bf;
typedef __attribute__((ext_vector_type(8)))  float  v8f;
typedef __attribute__((ext_vector_type(4)))  unsigned int u32x4;
typedef __attribute__((ext_vector_type(4)))  int i32x4;
typedef __attribute__((ext_vector_type(8)))  int i32x8;

#define LOG2E 1.44269504088896340736f

// ragged head bounds and per-head padded (to 32) column offsets (stride 544)
__device__ __constant__ int c_bnd[10]  = {0,7,21,49,105,161,217,273,357,441};
__device__ __constant__ int c_hoff[9]  = {0,32,64,96,160,224,288,352,448};

static __device__ inline v16bf ld_pair(const __bf16* lo, const __bf16* hi) {
  v8bf a = *(const v8bf*)lo;
  v8bf b = *(const v8bf*)hi;
  return __builtin_shufflevector(a, b, 0,1,2,3,4,5,6,7,8,9,10,11,12,13,14,15);
}
// A fragment (16x32 bf16, ISA 7.12.2): lane holds row (lane&15);
// VGPR0-3 = K[half*8 .. half*8+7], VGPR4-7 = K[16+half*8 .. 16+half*8+7]
static __device__ inline v16bf ld_fragA(const __bf16* rowbase, int half) {
  return ld_pair(rowbase + half*8, rowbase + 16 + half*8);
}
// B fragment (32x16 bf16): lane holds col (lane&15); K = half*8..half*8+7, 16+half*8..16+half*8+7
static __device__ inline v16bf ld_fragB(const __bf16* rowbase, int half) {
  return ld_pair(rowbase + half*8, rowbase + 16 + half*8);
}
static __device__ inline v8f wmma_bf16(v16bf a, v16bf b, v8f c) {
  return __builtin_amdgcn_wmma_f32_16x16x32_bf16(false, a, false, b, (short)0, c,
                                                 false, false);
}

// TDM: DMA a 128x32 bf16 tile (row stride 448) from global into LDS.
// D# pad feature inserts 4 DWORDs after every 16 DWORDs -> LDS row stride 40
// bf16, matching the aligned ds_load_b128 fragment layout.
static __device__ inline void tdm_load_tile_128x32(unsigned lds_off,
                                                   const __bf16* gsrc) {
  unsigned long long ga = (unsigned long long)(const void*)gsrc;
  u32x4 g0;
  g0[0] = 1u;                                    // count=1 (valid user D#)
  g0[1] = lds_off;                               // lds_addr (bytes)
  g0[2] = (unsigned)ga;                          // global_addr[31:0]
  g0[3] = (unsigned)((ga >> 32) & 0x01FFFFFFull) // global_addr[56:32]
        | (2u << 30);                            // type = 2 ("image")
  i32x8 g1;
  g1[0] = (1 << 16)                              // data_size = 2 bytes
        | (1 << 20)                              // pad_enable
        | (3 << 22)                              // pad_interval: 16 DWORDs
        | (3 << 25);                             // pad_amount:   4 DWORDs
  g1[1] = (448 & 0xffff) << 16;                  // tensor_dim0.lo16 (abar=0)
  g1[2] = (448 >> 16) | ((16384 & 0xffff) << 16);// dim0.hi16 | tensor_dim1.lo16
  g1[3] = (16384 >> 16) | (32 << 16);            // dim1.hi16 | tile_dim0 = 32
  g1[4] = 128;                                   // tile_dim1 = 128, tile_dim2 = 0
  g1[5] = 448;                                   // tensor_dim0_stride.lo32
  g1[6] = 0;                                     // stride0.hi16 | stride1.lo16
  g1[7] = 0;
  i32x4 zz = {0, 0, 0, 0};
#if __clang_major__ >= 23
  i32x8 z8 = {0, 0, 0, 0, 0, 0, 0, 0};
  __builtin_amdgcn_tensor_load_to_lds(g0, g1, zz, zz, z8, 0);
#else
  __builtin_amdgcn_tensor_load_to_lds(g0, g1, zz, zz, 0);
#endif
}

// ---------------- prep: fp32 -> bf16, pad 441 -> 448 ----------------
__global__ void prep_x_kernel(const float* __restrict__ X, __bf16* __restrict__ Xb) {
  int idx = blockIdx.x * 256 + threadIdx.x;       // over 16384*448
  int r = idx / 448, c = idx - r * 448;
  float v = (c < 441) ? X[(size_t)r * 441 + c] : 0.f;
  Xb[idx] = (__bf16)v;
}

__global__ void prep_w_kernel(const float* w0, const float* w1, const float* w2,
                              const float* w3, __bf16* o0, __bf16* o1, __bf16* o2,
                              __bf16* o3) {
  int which = blockIdx.y;
  const float* w = which == 0 ? w0 : which == 1 ? w1 : which == 2 ? w2 : w3;
  __bf16*      o = which == 0 ? o0 : which == 1 ? o1 : which == 2 ? o2 : o3;
  int idx = blockIdx.x * 256 + threadIdx.x;       // over 448*448
  int r = idx / 448, c = idx - r * 448;
  float v = (r < 441 && c < 441) ? w[(size_t)r * 441 + c] : 0.f;
  o[idx] = (__bf16)v;
}

// ---------------- WMMA GEMM: [16384 x 448(bf16)] @ [448 x 448(bf16)] ----------------
// MODE 0: out bf16, head-remapped padded layout (stride 544)   (Q, K)
// MODE 1: out bf16, transposed  out[col*16384 + row]           (V)
// MODE 2: out fp32, out[row*441 + col], col < 441              (final)
template <int MODE>
__global__ __launch_bounds__(256) void gemm_kernel(const __bf16* __restrict__ A,
                                                   const __bf16* __restrict__ W,
                                                   const float* __restrict__ bias,
                                                   void* __restrict__ outp) {
  __shared__ alignas(16) __bf16 sA[2][128 * 40]; // double-buffered TDM dest
  __shared__ alignas(16) __bf16 sB[64 * 40];     // 64 cols x 32 k (transposed)

  const int tid  = threadIdx.x;
  const int lane = tid & 31, wid = tid >> 5;
  const int half = lane >> 4, ln = lane & 15;
  const int wm = wid & 3, wn = wid >> 2;         // 4x2 wave grid, 32x32 per wave
  const int m0 = blockIdx.y * 128;
  const int n0 = blockIdx.x * 64;

  v8f z = {};
  v8f acc[2][2] = {{z, z}, {z, z}};

  const int bk = tid >> 3, bn0 = (tid & 7) * 8;

  // generic shared VA low 32 bits == LDS byte offset (LDS aperture layout)
  const unsigned ldsA0 = (unsigned)(unsigned long long)(const void*)&sA[0][0];
  const unsigned ldsA1 = (unsigned)(unsigned long long)(const void*)&sA[1][0];

  if (wid == 0) tdm_load_tile_128x32(ldsA0, A + (size_t)m0 * 448);

  int buf = 0;
  for (int kc = 0; kc < 448; kc += 32) {
    if (wid == 0) {
      if (kc + 32 < 448) {
        tdm_load_tile_128x32(buf ? ldsA0 : ldsA1,
                             A + (size_t)m0 * 448 + kc + 32);
        __builtin_amdgcn_s_wait_tensorcnt(1);  // tile kc resident (in-order TDM)
      } else {
        __builtin_amdgcn_s_wait_tensorcnt(0);
      }
    }
    // stage B tile 32x64 transposed -> sB[n][k] (all threads)
    {
      v8bf wv = *(const v8bf*)(W + (size_t)(kc + bk) * 448 + n0 + bn0);
#pragma unroll
      for (int j = 0; j < 8; ++j) sB[(bn0 + j) * 40 + bk] = wv[j];
    }
    __syncthreads();
    const __bf16* sa = sA[buf];
    v16bf a0 = ld_fragA(&sa[(wm * 32 + ln) * 40], half);
    v16bf a1 = ld_fragA(&sa[(wm * 32 + 16 + ln) * 40], half);
    v16bf b0 = ld_fragB(&sB[(wn * 32 + ln) * 40], half);
    v16bf b1 = ld_fragB(&sB[(wn * 32 + 16 + ln) * 40], half);
    acc[0][0] = wmma_bf16(a0, b0, acc[0][0]);
    acc[0][1] = wmma_bf16(a0, b1, acc[0][1]);
    acc[1][0] = wmma_bf16(a1, b0, acc[1][0]);
    acc[1][1] = wmma_bf16(a1, b1, acc[1][1]);
    __syncthreads();
    buf ^= 1;
  }

#pragma unroll
  for (int mi = 0; mi < 2; ++mi)
#pragma unroll
    for (int ni = 0; ni < 2; ++ni) {
      int mb  = m0 + wm * 32 + mi * 16;
      int nb  = n0 + wn * 32 + ni * 16;
      int col = nb + ln;
      float bv = (col < 441) ? bias[col] : 0.f;
      int pcol = 0;
      if (MODE == 0) {
        int hh = 0;
#pragma unroll
        for (int i = 1; i < 9; ++i)
          if (col >= c_bnd[i]) hh = i;
        pcol = c_hoff[hh] + col - c_bnd[hh];
      }
#pragma unroll
      for (int r = 0; r < 8; ++r) {
        int row = mb + r + 8 * half;          // C layout: M = r + 8*half
        float v = acc[mi][ni][r] + bv;
        if (MODE == 0) {
          if (col < 441) ((__bf16*)outp)[(size_t)row * 544 + pcol] = (__bf16)v;
        } else if (MODE == 1) {
          ((__bf16*)outp)[(size_t)col * 16384 + row] = (__bf16)v;
        } else {
          if (col < 441) ((float*)outp)[(size_t)row * 441 + col] = v;
        }
      }
    }
}

// ---------------- flash attention, one wave per 16 q-rows ----------------
__global__ __launch_bounds__(256) void attn_kernel(const __bf16* __restrict__ Qp,
                                                   const __bf16* __restrict__ Kp,
                                                   const __bf16* __restrict__ Vt,
                                                   __bf16* __restrict__ Ctx) {
  const int b = blockIdx.z, h = blockIdx.y;
  const int s = c_bnd[h], e = c_bnd[h + 1];
  const int d = e - s;
  const int hoff = c_hoff[h];
  const int nkc  = (d + 31) >> 5;   // 32-wide K chunks (1..3)
  const int nd16 = (d + 15) >> 4;   // 16-wide output dim tiles (1..6)
  const int tid = threadIdx.x, lane = tid & 31, wid = tid >> 5;
  const int half = lane >> 4, ln = lane & 15;
  const int m0 = (blockIdx.x * 8 + wid) * 16;   // q-tile within batch

  __shared__ alignas(16) __bf16 sP[8 * 16 * 40];  // per-wave P transpose slab
  __bf16* sp = &sP[wid * 16 * 40];

  // Q fragments (padded layout: chunks beyond head are zero)
  const __bf16* qrow = Qp + (size_t)(b * 1024 + m0 + ln) * 544 + hoff;
  v16bf qf[3];
#pragma unroll
  for (int c = 0; c < 3; ++c)
    if (c < nkc) qf[c] = ld_fragA(qrow + c * 32, half);

  const float e2 = rsqrtf((float)d) * LOG2E;   // fold 1/sqrt(d) into exp2 domain
  float mrow[8], lrow[8];
#pragma unroll
  for (int r = 0; r < 8; ++r) { mrow[r] = -3.0e38f; lrow[r] = 0.f; }
  v8f z = {};
  v8f acc[6] = {z, z, z, z, z, z};

  for (int kt = 0; kt < 32; ++kt) {   // 32 keys per iteration
    const int krow0 = b * 1024 + kt * 32;
    if (kt < 31)
      __builtin_prefetch(Kp + (size_t)(krow0 + 32 + ln) * 544 + hoff, 0, 0);

    // scores: two 16x16 tiles (keys 0-15, 16-31)
    v8f s0 = z, s1 = z;
#pragma unroll
    for (int c = 0; c < 3; ++c)
      if (c < nkc) {
        const __bf16* kb = Kp + (size_t)(krow0 + ln) * 544 + hoff + c * 32;
        v16bf bk0 = ld_fragB(kb, half);
        v16bf bk1 = ld_fragB(kb + (size_t)16 * 544, half);
        s0 = wmma_bf16(qf[c], bk0, s0);
        s1 = wmma_bf16(qf[c], bk1, s1);
      }

    // online softmax: rows live in 16-lane halves; xor 1/2/4/8 reduces per-half
#pragma unroll
    for (int r = 0; r < 8; ++r) {
      float t0 = s0[r] * e2, t1 = s1[r] * e2;
      float mx = fmaxf(t0, t1);
      mx = fmaxf(mx, __shfl_xor(mx, 1));
      mx = fmaxf(mx, __shfl_xor(mx, 2));
      mx = fmaxf(mx, __shfl_xor(mx, 4));
      mx = fmaxf(mx, __shfl_xor(mx, 8));
      float mnew = fmaxf(mrow[r], mx);
      float al = exp2f(mrow[r] - mnew);
      float p0 = exp2f(t0 - mnew);
      float p1 = exp2f(t1 - mnew);
      float ps = p0 + p1;
      ps += __shfl_xor(ps, 1);
      ps += __shfl_xor(ps, 2);
      ps += __shfl_xor(ps, 4);
      ps += __shfl_xor(ps, 8);
      lrow[r] = lrow[r] * al + ps;
      mrow[r] = mnew;
#pragma unroll
      for (int nt = 0; nt < 6; ++nt)
        if (nt < nd16) acc[nt][r] *= al;
      // P (C layout) -> LDS row-major for A-fragment reload
      sp[(r + 8 * half) * 40 + ln]      = (__bf16)p0;
      sp[(r + 8 * half) * 40 + 16 + ln] = (__bf16)p1;
    }
    __syncthreads();
    v16bf pa = ld_fragA(&sp[ln * 40], half);   // P as 16x32 A fragment
#pragma unroll
    for (int nt = 0; nt < 6; ++nt)
      if (nt < nd16) {
        int gd = s + nt * 16 + ln;
        gd = gd > 447 ? 447 : gd;              // clamp pad cols in-bounds
        v16bf bv = ld_fragB(Vt + (size_t)gd * 16384 + krow0, half);
        acc[nt] = wmma_bf16(pa, bv, acc[nt]);
      }
    __syncthreads();
  }

#pragma unroll
  for (int r = 0; r < 8; ++r) {
    float inv = 1.0f / lrow[r];
    size_t row = (size_t)(b * 1024 + m0 + r + 8 * half);
#pragma unroll
    for (int nt = 0; nt < 6; ++nt)
      if (nt < nd16) {
        int col = s + nt * 16 + ln;
        if (col < e) Ctx[row * 448 + col] = (__bf16)(acc[nt][r] * inv);
      }
  }
}

// ---------------- launch ----------------
extern "C" void kernel_launch(void* const* d_in, const int* in_sizes, int n_in,
                              void* d_out, int out_size, void* d_ws, size_t ws_size,
                              hipStream_t stream) {
  (void)in_sizes; (void)n_in; (void)out_size; (void)ws_size;
  const float* X  = (const float*)d_in[0];
  const float* Wq = (const float*)d_in[1];
  const float* bq = (const float*)d_in[2];
  const float* Wk = (const float*)d_in[3];
  const float* bk = (const float*)d_in[4];
  const float* Wv = (const float*)d_in[5];
  const float* bv = (const float*)d_in[6];
  const float* Wo = (const float*)d_in[7];
  const float* bo = (const float*)d_in[8];
  float* out = (float*)d_out;

  char* p = (char*)d_ws;
  const size_t XB = (size_t)16384 * 448 * 2;   // bf16 [16384][448]
  const size_t WB = (size_t)448 * 448 * 2;     // bf16 [448][448]
  const size_t QP = (size_t)16384 * 544 * 2;   // bf16 per-head-padded [16384][544]
  __bf16* Xb  = (__bf16*)p; p += XB;
  __bf16* Wqb = (__bf16*)p; p += WB;
  __bf16* Wkb = (__bf16*)p; p += WB;
  __bf16* Wvb = (__bf16*)p; p += WB;
  __bf16* Wob = (__bf16*)p; p += WB;
  __bf16* Qp  = (__bf16*)p; p += QP;
  __bf16* Kp  = (__bf16*)p; p += QP;
  __bf16* Vt  = (__bf16*)p; p += XB;           // bf16 transposed [448][16384]
  __bf16* Ctx = (__bf16*)p; p += XB;

  hipMemsetAsync(Qp, 0, QP, stream);           // zero pad gaps between heads
  hipMemsetAsync(Kp, 0, QP, stream);
  hipMemsetAsync(Ctx, 0, XB, stream);          // zero cols 441..447

  prep_x_kernel<<<16384 * 448 / 256, 256, 0, stream>>>(X, Xb);
  prep_w_kernel<<<dim3(448 * 448 / 256, 4), 256, 0, stream>>>(Wq, Wk, Wv, Wo, Wqb,
                                                              Wkb, Wvb, Wob);

  dim3 ggrid(7, 128);  // N tiles of 64, M tiles of 128
  gemm_kernel<0><<<ggrid, 256, 0, stream>>>(Xb, Wqb, bq, Qp);
  gemm_kernel<0><<<ggrid, 256, 0, stream>>>(Xb, Wkb, bk, Kp);
  gemm_kernel<1><<<ggrid, 256, 0, stream>>>(Xb, Wvb, bv, Vt);

  attn_kernel<<<dim3(8, 9, 16), 256, 0, stream>>>(Qp, Kp, Vt, Ctx);

  gemm_kernel<2><<<ggrid, 256, 0, stream>>>(Ctx, Wob, bo, (void*)out);
}
